// S4Layer_25022479466510
// MI455X (gfx1250) — matrix operand order repaired and checked by reading, not compiled, not run on hardware
//
#include <hip/hip_runtime.h>
#include <hip/hip_bf16.h>
#include <math.h>

#define B_   8
#define L_   4096
#define H_   256
#define N2_  32
#define NL_  4

typedef __attribute__((ext_vector_type(16))) __bf16 v16bf;
typedef __attribute__((ext_vector_type(8)))  __bf16 v8bf;
typedef __attribute__((ext_vector_type(8)))  float  v8f;
typedef __attribute__((ext_vector_type(4)))  int    v4i;

#if defined(__has_builtin)
#if __has_builtin(__builtin_amdgcn_global_load_async_to_lds_b128) && \
    __has_builtin(__builtin_amdgcn_s_wait_asynccnt)
#define USE_ASYNC_LDS 1
#endif
#endif
#ifndef USE_ASYNC_LDS
#define USE_ASYNC_LDS 0
#endif

#if USE_ASYNC_LDS
typedef __attribute__((address_space(1))) v4i* gv4i_p;   // global b128 payload ptr
typedef __attribute__((address_space(3))) v4i* lv4i_p;   // LDS b128 payload ptr
#endif

// ---------------------------------------------------------------------------
// Input transpose: x (B,L,H) f32 -> u (B,H,L) f32, LDS-tiled 32x32
// ---------------------------------------------------------------------------
__global__ void k_transpose_in(const float* __restrict__ x, float* __restrict__ u) {
  __shared__ float tile[32][33];
  const int b  = blockIdx.z;
  const int l0 = blockIdx.x * 32;
  const int h0 = blockIdx.y * 32;
  for (int r = threadIdx.y; r < 32; r += 8)
    tile[r][threadIdx.x] = x[((size_t)b * L_ + (l0 + r)) * H_ + h0 + threadIdx.x];
  __syncthreads();
  for (int r = threadIdx.y; r < 32; r += 8)
    u[((size_t)b * H_ + (h0 + r)) * L_ + l0 + threadIdx.x] = tile[threadIdx.x][r];
}

// ---------------------------------------------------------------------------
// Per-mode ZOH discretization: Abar = exp(dt*A), Cm = 2*C*(exp(dt*A)-1)/A
// ---------------------------------------------------------------------------
__global__ void k_params(const float* __restrict__ log_dt,
                         const float* __restrict__ log_A_real,
                         const float* __restrict__ A_imag,
                         const float* __restrict__ C_re,
                         const float* __restrict__ C_im,
                         float* __restrict__ Ab_re, float* __restrict__ Ab_im,
                         float* __restrict__ Cm_re, float* __restrict__ Cm_im) {
  const int idx = blockIdx.x * blockDim.x + threadIdx.x;   // over NL*H*N2
  if (idx >= NL_ * H_ * N2_) return;
  const int i = idx / (H_ * N2_);
  const int h = (idx / N2_) % H_;
  const float dt = expf(log_dt[i * H_ + h]);
  const float Ar = -expf(log_A_real[idx]);
  const float Ai = A_imag[idx];
  const float xr = dt * Ar, xi = dt * Ai;
  const float e  = expf(xr);
  const float abr = e * cosf(xi);
  const float abi = e * sinf(xi);
  const float nr = abr - 1.0f, ni = abi;
  const float inv = 1.0f / (Ar * Ar + Ai * Ai);
  const float dr = (nr * Ar + ni * Ai) * inv;
  const float di = (ni * Ar - nr * Ai) * inv;
  const float cr = C_re[idx], ci = C_im[idx];
  Ab_re[idx] = abr;
  Ab_im[idx] = abi;
  Cm_re[idx] = 2.0f * (cr * dr - ci * di);   // fold the 2*Re(...) factor
  Cm_im[idx] = 2.0f * (cr * di + ci * dr);
}

// ---------------------------------------------------------------------------
// Wout f32 -> bf16
// ---------------------------------------------------------------------------
__global__ void k_w2bf(const float* __restrict__ W, __hip_bfloat16* __restrict__ Wb, int n) {
  const int i = blockIdx.x * blockDim.x + threadIdx.x;
  if (i < n) Wb[i] = __float2bfloat16(W[i]);
}

// ---------------------------------------------------------------------------
// Diagonal SSM recurrence + D*u + exact GELU.
// One wave per (b,h); lane n holds complex state of mode n (N2 == wave32).
// Writes g (B,L,H) bf16 for the WMMA channel-mix that follows.
// ---------------------------------------------------------------------------
__global__ void k_scan(const float* __restrict__ u,
                       const float* __restrict__ Ab_re, const float* __restrict__ Ab_im,
                       const float* __restrict__ Cm_re, const float* __restrict__ Cm_im,
                       const float* __restrict__ Dv,
                       __hip_bfloat16* __restrict__ g, int layer) {
  const int wave = (blockIdx.x * blockDim.x + threadIdx.x) >> 5;
  const int lane = threadIdx.x & 31;
  const int b = wave / H_;
  const int h = wave % H_;
  const int pidx = (layer * H_ + h) * N2_ + lane;
  const float ar = Ab_re[pidx], ai = Ab_im[pidx];
  const float cr = Cm_re[pidx], ci = Cm_im[pidx];
  const float dco = Dv[layer * H_ + h];
  const float4* up = reinterpret_cast<const float4*>(u + ((size_t)b * H_ + h) * L_);
  __hip_bfloat16* gp = g + (size_t)b * L_ * H_ + h;
  float sr = 0.0f, si = 0.0f;
  for (int l4 = 0; l4 < L_ / 4; ++l4) {
    const float4 uv = up[l4];
    __builtin_prefetch(up + l4 + 128, 0, 1);     // global_prefetch_b8, ~2KB ahead
    const float uu[4] = {uv.x, uv.y, uv.z, uv.w};
#pragma unroll
    for (int j = 0; j < 4; ++j) {
      const float nsr = ar * sr - ai * si + uu[j];
      const float nsi = ar * si + ai * sr;
      sr = nsr; si = nsi;
      float p = cr * sr - ci * si;               // Re(Cm * s), factor 2 folded in
#pragma unroll
      for (int off = 16; off > 0; off >>= 1)
        p += __shfl_xor(p, off, 32);
      if (lane == 0) {
        const float y  = p + dco * uu[j];
        const float yg = 0.5f * y * (1.0f + erff(y * 0.70710678118654752f));
        gp[(size_t)(l4 * 4 + j) * H_] = __float2bfloat16(yg);
      }
    }
  }
}

// ---------------------------------------------------------------------------
// Channel mix: z2 = Wout @ g + bout, then GLU a*sigmoid(b).
// One block (8 waves) owns one (b, 16-wide l strip). The streaming operand
// (activation strip g[b, l0:l0+16, :], 8KB contiguous) is staged ONCE into
// LDS via GLOBAL_LOAD_ASYNC_TO_LDS_B128 (+ s_wait_asynccnt), then each wave
// computes 2 ot-pairs (o and o+H share the B fragment) -> 4 v_wmma per
// K-step, 32 static WMMA. Weights (256KB/layer) stay resident in WGP$/L2.
// ---------------------------------------------------------------------------
__global__ void k_gemm_glu(const __hip_bfloat16* __restrict__ g,
                           const __hip_bfloat16* __restrict__ Wb,
                           const float* __restrict__ bout,
                           float* __restrict__ z, int layer) {
  __shared__ __align__(16) __hip_bfloat16 Bs[16 * H_];   // 8KB strip
  const int TL = L_ / 16;                 // 256 l-tiles
  const int b  = blockIdx.x / TL;
  const int l0 = (blockIdx.x % TL) * 16;
  const int tid  = threadIdx.x;           // 0..255
  const int lane = tid & 31;
  const int wv   = tid >> 5;              // 0..7

  // --- stage B strip: g[b, l0.., :] is 4096 contiguous bf16 (8KB) ---
  const __hip_bfloat16* gsrc = g + ((size_t)b * L_ + l0) * H_;
#if USE_ASYNC_LDS
  {
    gv4i_p g0  = (gv4i_p)(gsrc + tid * 16);
    gv4i_p g1  = (gv4i_p)(gsrc + tid * 16 + 8);
    lv4i_p l0p = (lv4i_p)(&Bs[tid * 16]);
    lv4i_p l1p = (lv4i_p)(&Bs[tid * 16 + 8]);
    __builtin_amdgcn_global_load_async_to_lds_b128(g0, l0p, 0, 0);
    __builtin_amdgcn_global_load_async_to_lds_b128(g1, l1p, 0, 0);
    __builtin_amdgcn_s_wait_asynccnt(0);
  }
  __syncthreads();
#else
  {
    const float4* s = reinterpret_cast<const float4*>((const void*)gsrc);
    float4* d = reinterpret_cast<float4*>((void*)Bs);
    d[tid]       = s[tid];
    d[tid + 256] = s[tid + 256];
  }
  __syncthreads();
#endif

  const int hs = lane >> 4;               // lane half-select
  const int lm = lane & 15;
  const int akb = hs * 8;                 // A frag: K = akb..+7 and akb+16..+23
  const int bkb = hs * 8;                 // B frag: K = bkb..+7 and bkb+16..+23

  // wave wv covers ot-pairs {wv, wv+8} of the 16 GLU-half row tiles
  const __hip_bfloat16* Wa[2];
  const __hip_bfloat16* Wc[2];
  int h0t[2];
#pragma unroll
  for (int t = 0; t < 2; ++t) {
    h0t[t] = (wv + 8 * t) * 16;
    Wa[t] = Wb + ((size_t)layer * 2 * H_ + h0t[t] + lm) * H_;
    Wc[t] = Wa[t] + (size_t)H_ * H_;      // GLU gate rows o+H
  }
  const __hip_bfloat16* Bp = &Bs[lm * H_];

  v8f c0[2] = {v8f{}, v8f{}};
  v8f c1[2] = {v8f{}, v8f{}};
#pragma unroll
  for (int k0 = 0; k0 < H_; k0 += 32) {
    const v8bf blo = *reinterpret_cast<const v8bf*>((const void*)(Bp + k0 + bkb));
    const v8bf bhi = *reinterpret_cast<const v8bf*>((const void*)(Bp + k0 + bkb + 16));
    v16bf bm;
#pragma unroll
    for (int q = 0; q < 8; ++q) { bm[q] = blo[q]; bm[8 + q] = bhi[q]; }
#pragma unroll
    for (int t = 0; t < 2; ++t) {
      const v8bf a0lo = *reinterpret_cast<const v8bf*>((const void*)(Wa[t] + k0 + akb));
      const v8bf a0hi = *reinterpret_cast<const v8bf*>((const void*)(Wa[t] + k0 + akb + 16));
      const v8bf a1lo = *reinterpret_cast<const v8bf*>((const void*)(Wc[t] + k0 + akb));
      const v8bf a1hi = *reinterpret_cast<const v8bf*>((const void*)(Wc[t] + k0 + akb + 16));
      v16bf a0, a1;
#pragma unroll
      for (int q = 0; q < 8; ++q) {
        a0[q] = a0lo[q]; a0[8 + q] = a0hi[q];
        a1[q] = a1lo[q]; a1[8 + q] = a1hi[q];
      }
      c0[t] = __builtin_amdgcn_wmma_f32_16x16x32_bf16(false, a0, false, bm, (short)0, c0[t], false, false);
      c1[t] = __builtin_amdgcn_wmma_f32_16x16x32_bf16(false, a1, false, bm, (short)0, c1[t], false, false);
    }
  }

  // Epilogue: bias + GLU. C layout: m = r + 8*hs, n = lm.
  const float* bo = bout + (size_t)layer * 2 * H_;
#pragma unroll
  for (int t = 0; t < 2; ++t) {
    float zz[8];
#pragma unroll
    for (int r = 0; r < 8; ++r) {
      const int hh = h0t[t] + r + hs * 8;
      const float av = c0[t][r] + bo[hh];
      const float bv = c1[t][r] + bo[H_ + hh];
      zz[r] = av * (1.0f / (1.0f + expf(-bv)));
    }
    float* zp = z + ((size_t)b * L_ + (l0 + lm)) * H_ + h0t[t] + hs * 8;
    *reinterpret_cast<float4*>(zp)     = make_float4(zz[0], zz[1], zz[2], zz[3]);
    *reinterpret_cast<float4*>(zp + 4) = make_float4(zz[4], zz[5], zz[6], zz[7]);
  }
}

// ---------------------------------------------------------------------------
// Residual + channel LayerNorm. One wave per (b,l), 8 channels per lane.
// Writes u (B,H,L) for next layer, or d_out (B,L,H) on last layer.
// ---------------------------------------------------------------------------
__global__ void k_resln(const float* __restrict__ z, float* u,
                        const float* __restrict__ lng, const float* __restrict__ lnb,
                        float* outBLH, int layer, int last) {
  const int wave = (blockIdx.x * blockDim.x + threadIdx.x) >> 5;
  const int lane = threadIdx.x & 31;
  const int b = wave / L_;
  const int l = wave % L_;
  const float* zp = z + ((size_t)b * L_ + l) * H_;
  float* up = u + (size_t)b * H_ * L_ + l;
  float v[8];
  float s = 0.0f, s2 = 0.0f;
#pragma unroll
  for (int j = 0; j < 8; ++j) {
    const int h = lane + 32 * j;
    const float t = zp[h] + up[(size_t)h * L_];
    v[j] = t; s += t; s2 += t * t;
  }
#pragma unroll
  for (int off = 16; off > 0; off >>= 1) {
    s  += __shfl_xor(s,  off, 32);
    s2 += __shfl_xor(s2, off, 32);
  }
  const float mean = s * (1.0f / H_);
  const float var  = s2 * (1.0f / H_) - mean * mean;
  const float inv  = rsqrtf(var + 1e-5f);
#pragma unroll
  for (int j = 0; j < 8; ++j) {
    const int h = lane + 32 * j;
    const float o = (v[j] - mean) * inv * lng[layer * H_ + h] + lnb[layer * H_ + h];
    if (last) outBLH[((size_t)b * L_ + l) * H_ + h] = o;
    else      up[(size_t)h * L_] = o;
  }
}

// ---------------------------------------------------------------------------
extern "C" void kernel_launch(void* const* d_in, const int* in_sizes, int n_in,
                              void* d_out, int out_size, void* d_ws, size_t ws_size,
                              hipStream_t stream) {
  const float* x          = (const float*)d_in[0];
  const float* log_dt     = (const float*)d_in[1];
  const float* log_A_real = (const float*)d_in[2];
  const float* A_imag     = (const float*)d_in[3];
  const float* C_re       = (const float*)d_in[4];
  const float* C_im       = (const float*)d_in[5];
  const float* Dv         = (const float*)d_in[6];
  const float* Wout       = (const float*)d_in[7];
  const float* bout       = (const float*)d_in[8];
  const float* ln_g       = (const float*)d_in[9];
  const float* ln_b       = (const float*)d_in[10];
  float* out = (float*)d_out;

  char* ws = (char*)d_ws;
  size_t off = 0;
  float* u = (float*)(ws + off);              off += (size_t)B_ * H_ * L_ * sizeof(float);
  __hip_bfloat16* g = (__hip_bfloat16*)(ws + off); off += (size_t)B_ * L_ * H_ * sizeof(__hip_bfloat16);
  float* z = (float*)(ws + off);              off += (size_t)B_ * L_ * H_ * sizeof(float);
  __hip_bfloat16* Wb = (__hip_bfloat16*)(ws + off); off += (size_t)NL_ * 2 * H_ * H_ * sizeof(__hip_bfloat16);
  const size_t np = (size_t)NL_ * H_ * N2_;
  float* Abre = (float*)(ws + off); off += np * sizeof(float);
  float* Abim = (float*)(ws + off); off += np * sizeof(float);
  float* Cmre = (float*)(ws + off); off += np * sizeof(float);
  float* Cmim = (float*)(ws + off); off += np * sizeof(float);

  // One-time prep
  dim3 tb(32, 8), tg(L_ / 32, H_ / 32, B_);
  k_transpose_in<<<tg, tb, 0, stream>>>(x, u);
  k_params<<<(NL_ * H_ * N2_ + 255) / 256, 256, 0, stream>>>(
      log_dt, log_A_real, A_imag, C_re, C_im, Abre, Abim, Cmre, Cmim);
  const int nw = NL_ * 2 * H_ * H_;
  k_w2bf<<<(nw + 255) / 256, 256, 0, stream>>>(Wout, Wb, nw);

  const int scan_blocks = (B_ * H_ * 32) / 256;   // 256
  const int gemm_blocks = B_ * (L_ / 16);         // 2048 (one block per l-strip)
  const int ln_blocks   = (B_ * L_ * 32) / 256;   // 4096
  for (int i = 0; i < NL_; ++i) {
    k_scan<<<scan_blocks, 256, 0, stream>>>(u, Abre, Abim, Cmre, Cmim, Dv, g, i);
    k_gemm_glu<<<gemm_blocks, 256, 0, stream>>>(g, Wb, bout, z, i);
    k_resln<<<ln_blocks, 256, 0, stream>>>(z, u, ln_g, ln_b, out, i, i == NL_ - 1 ? 1 : 0);
  }
}
